// PointNetReg_77532749627398
// MI455X (gfx1250) — matrix operand order repaired and checked by reading, not compiled, not run on hardware
//
#include <hip/hip_runtime.h>
#include <math.h>

typedef __attribute__((ext_vector_type(16))) _Float16 v16h;
typedef __attribute__((ext_vector_type(8)))  _Float16 v8h;
typedef __attribute__((ext_vector_type(8)))  float    v8f;

#define CDIV(a, b) (((a) + (b) - 1) / (b))
static inline int align16(int x) { return (x + 15) & ~15; }
static inline int align32(int x) { return (x + 31) & ~31; }

// ---------------------------------------------------------------------------
// xyz (B,3,N) -> coords (B,N,3)
// ---------------------------------------------------------------------------
__global__ void transpose_xyz_kernel(const float* __restrict__ xyz,
                                     float* __restrict__ coords) {
  int i = blockIdx.x * blockDim.x + threadIdx.x;   // over 16*4096
  if (i >= 16 * 4096) return;
  int b = i / 4096, n = i % 4096;
  coords[(size_t)i * 3 + 0] = xyz[((size_t)b * 3 + 0) * 4096 + n];
  coords[(size_t)i * 3 + 1] = xyz[((size_t)b * 3 + 1) * 4096 + n];
  coords[(size_t)i * 3 + 2] = xyz[((size_t)b * 3 + 2) * 4096 + n];
}

// ---------------------------------------------------------------------------
// Farthest point sampling: one block per batch, dist array in LDS.
// Emits `far` BEFORE the update (lax.scan semantics); argmax keeps first
// (lowest-index) maximum on ties.
// ---------------------------------------------------------------------------
#define FPS_T 256
__global__ void fps_kernel(const float* __restrict__ coords, int n, int npoint,
                           int* __restrict__ out_idx) {
  int b = blockIdx.x;
  const float* P = coords + (size_t)b * n * 3;
  __shared__ float dist[4096];
  __shared__ float rv[FPS_T];
  __shared__ int   ri[FPS_T];
  __shared__ int   s_far;
  int t = threadIdx.x;
  for (int i = t; i < n; i += FPS_T) dist[i] = 1e10f;
  if (t == 0) s_far = 0;
  __syncthreads();
  for (int it = 0; it < npoint; ++it) {
    int far = s_far;
    if (t == 0) out_idx[b * npoint + it] = far;
    float cx = P[far * 3 + 0], cy = P[far * 3 + 1], cz = P[far * 3 + 2];
    float bestv = -1.f; int besti = 0;
    for (int i = t; i < n; i += FPS_T) {
      float dx = P[i * 3 + 0] - cx;
      float dy = P[i * 3 + 1] - cy;
      float dz = P[i * 3 + 2] - cz;
      float d = dx * dx + dy * dy + dz * dz;
      float dm = fminf(dist[i], d);
      dist[i] = dm;
      if (dm > bestv) { bestv = dm; besti = i; }
    }
    rv[t] = bestv; ri[t] = besti;
    __syncthreads();
    for (int s = FPS_T / 2; s > 0; s >>= 1) {
      if (t < s) {
        if (rv[t + s] > rv[t] || (rv[t + s] == rv[t] && ri[t + s] < ri[t])) {
          rv[t] = rv[t + s]; ri[t] = ri[t + s];
        }
      }
      __syncthreads();
    }
    if (t == 0) s_far = ri[0];
    __syncthreads();
  }
}

// ---------------------------------------------------------------------------
// Ball query: sort-then-truncate == first 32 ascending indices with d^2<=r^2;
// remaining slots replicate the first hit. Also emits new_xyz.
// ---------------------------------------------------------------------------
__global__ void ballquery_kernel(const float* __restrict__ coords,
                                 const int* __restrict__ fps_idx, float r2,
                                 int n, int npoint,
                                 float* __restrict__ new_xyz,
                                 int* __restrict__ group_idx) {
  int q = blockIdx.x * blockDim.x + threadIdx.x;
  if (q >= 16 * npoint) return;
  int b = q / npoint, s = q % npoint;
  const float* P = coords + (size_t)b * n * 3;
  int ctr = fps_idx[b * npoint + s];
  float qx = P[ctr * 3], qy = P[ctr * 3 + 1], qz = P[ctr * 3 + 2];
  new_xyz[(size_t)q * 3 + 0] = qx;
  new_xyz[(size_t)q * 3 + 1] = qy;
  new_xyz[(size_t)q * 3 + 2] = qz;
  int* gi = group_idx + (size_t)q * 32;
  int cnt = 0, first = -1;
  for (int j = 0; j < n && cnt < 32; ++j) {
    float dx = P[j * 3] - qx, dy = P[j * 3 + 1] - qy, dz = P[j * 3 + 2] - qz;
    if (dx * dx + dy * dy + dz * dz <= r2) {
      if (first < 0) first = j;
      gi[cnt++] = j;
    }
  }
  if (first < 0) first = n - 1;   // JAX OOB gather clamps
  for (; cnt < 32; ++cnt) gi[cnt] = first;
}

// ---------------------------------------------------------------------------
// Grouping: writes padded f16 GEMM input directly.
// row=(b,s,k): out[row,:] = concat(xyz[j]-new_xyz, feats[j], zeros to Kpad)
// ---------------------------------------------------------------------------
__global__ void group_kernel(const float* __restrict__ coords,
                             const float* __restrict__ feats,
                             const float* __restrict__ new_xyz,
                             const int* __restrict__ group_idx,
                             int n, int npoint, int Cprev, int Kpad,
                             _Float16* __restrict__ out) {
  int row = blockIdx.x * blockDim.x + threadIdx.x;
  int M = 16 * npoint * 32;
  if (row >= M) return;
  int k = row & 31;
  int q = row >> 5;             // b*npoint + s
  int b = q / npoint;
  int j = group_idx[(size_t)q * 32 + k];
  const float* P  = coords + ((size_t)b * n + j) * 3;
  const float* nx = new_xyz + (size_t)q * 3;
  _Float16* o = out + (size_t)row * Kpad;
  o[0] = (_Float16)(P[0] - nx[0]);
  o[1] = (_Float16)(P[1] - nx[1]);
  o[2] = (_Float16)(P[2] - nx[2]);
  const float* f = feats + ((size_t)b * n + j) * Cprev;
  for (int c = 0; c < Cprev; ++c) o[3 + c] = (_Float16)f[c];
  for (int c = 3 + Cprev; c < Kpad; ++c) o[c] = (_Float16)0.f;
}

// ---------------------------------------------------------------------------
// Weight pack: w (N x K f32) -> wp (Npad x Kpad f16), zero padded both dims.
// ---------------------------------------------------------------------------
__global__ void pack_w_kernel(const float* __restrict__ w,
                              _Float16* __restrict__ wp,
                              int N, int K, int Npad, int Kpad) {
  int i = blockIdx.x * blockDim.x + threadIdx.x;
  if (i >= Npad * Kpad) return;
  int nn = i / Kpad, kk = i % Kpad;
  wp[i] = (_Float16)((nn < N && kk < K) ? w[(size_t)nn * K + kk] : 0.f);
}

// ---------------------------------------------------------------------------
// WMMA GEMM: out[m,n] = sum_k A16[m,k]*Wp[n,k] + bias[n]
// A16: M x Kpad f16 (zero-padded), Wp: Npad x Kpad f16 (zero-padded).
// One 16x16 tile per wave; unconditional vector loads, no inner-loop guards.
// Fragment layouts used here (contiguous k, same order for A and B):
//   A 16x32: lanes<16 K 0..15; lanes>=16 K 16..31
//   B 32x16: lanes<16 K 0..15; lanes>=16 K 16..31
//   C/D    : lanes<16 rows 0..7; lanes>=16 rows 8..15; col = lane&15
// ---------------------------------------------------------------------------
__global__ void wmma_nt_kernel(const _Float16* __restrict__ A16,
                               const _Float16* __restrict__ Wp,
                               const float* __restrict__ bias,
                               float* __restrict__ out,
                               int M, int N, int Kpad) {
  int wave = (blockIdx.x * blockDim.x + threadIdx.x) >> 5;
  int lane = threadIdx.x & 31;
  int tilesN = (N + 15) >> 4;
  int tileM = (wave / tilesN) << 4;
  int tileN = (wave % tilesN) << 4;
  if (tileM >= M) return;
  int h   = lane >> 4;
  int l15 = lane & 15;
  const _Float16* Arow = A16 + (size_t)(tileM + l15) * Kpad + (h << 4);
  const _Float16* Wrow = Wp  + (size_t)(tileN + l15) * Kpad + (h << 4);
  v8f acc = {};
  for (int k0 = 0; k0 < Kpad; k0 += 32) {
    v8h  alo = *(const v8h*)(Arow + k0);        // K h*16+0..7   (16B aligned)
    v8h  ahi = *(const v8h*)(Arow + k0 + 8);    // K h*16+8..15  (16B aligned)
    v16h b   = *(const v16h*)(Wrow + k0);       // K h*16+0..15  (32B aligned)
    v16h a   = __builtin_shufflevector(alo, ahi, 0, 1, 2, 3, 4, 5, 6, 7,
                                       8, 9, 10, 11, 12, 13, 14, 15);
    acc = __builtin_amdgcn_wmma_f32_16x16x32_f16(
        false, a, false, b, (short)0, acc, false, false);
  }
  int colB = tileN + l15;
  if (colB < N) {
    float bv = bias[colB];
#pragma unroll
    for (int r = 0; r < 8; ++r)
      out[(size_t)(tileM + (h << 3) + r) * N + colB] = acc[r] + bv;
  }
}

// ---------------------------------------------------------------------------
// Deterministic per-channel BN stats (fixed-order tree reduction, no atomics)
// ---------------------------------------------------------------------------
__global__ void bn_stats_kernel(const float* __restrict__ x, int M, int N,
                                const float* __restrict__ g,
                                const float* __restrict__ beta,
                                float* __restrict__ scale,
                                float* __restrict__ shift) {
  int c = blockIdx.x;
  int t = threadIdx.x;
  __shared__ float ss[256], sq[256];
  float s = 0.f, s2 = 0.f;
  for (int m = t; m < M; m += 256) {
    float v = x[(size_t)m * N + c];
    s += v; s2 += v * v;
  }
  ss[t] = s; sq[t] = s2;
  __syncthreads();
  for (int k = 128; k > 0; k >>= 1) {
    if (t < k) { ss[t] += ss[t + k]; sq[t] += sq[t + k]; }
    __syncthreads();
  }
  if (t == 0) {
    float mu  = ss[0] / (float)M;
    float var = sq[0] / (float)M - mu * mu;
    float sc  = g[c] * rsqrtf(var + 1e-5f);
    scale[c] = sc;
    shift[c] = beta[c] - mu * sc;
  }
}

// ---------------------------------------------------------------------------
// Fused BN+ReLU: updates f32 tensor in place (for stats-free consumers like
// max-pool) and emits the next GEMM's zero-padded f16 input.
// Launched over M*Kpad elements.
// ---------------------------------------------------------------------------
__global__ void bn_relu_f16_kernel(float* __restrict__ x, size_t totalPad,
                                   int N, int Kpad,
                                   const float* __restrict__ scale,
                                   const float* __restrict__ shift,
                                   _Float16* __restrict__ out16) {
  size_t i = (size_t)blockIdx.x * blockDim.x + threadIdx.x;
  if (i >= totalPad) return;
  int c = (int)(i % (size_t)Kpad);
  size_t m = i / (size_t)Kpad;
  float v = 0.f;
  if (c < N) {
    v = x[m * N + c] * scale[c] + shift[c];
    v = fmaxf(v, 0.f);
    x[m * N + c] = v;
  }
  out16[i] = (_Float16)v;
}

// ---------------------------------------------------------------------------
// Max-pool over the 32 samples of each group (reads normalized f32)
// ---------------------------------------------------------------------------
__global__ void maxpool_kernel(const float* __restrict__ x, int S, int C,
                               float* __restrict__ out) {
  int i = blockIdx.x * blockDim.x + threadIdx.x;   // over 16*S*C
  if (i >= 16 * S * C) return;
  int c = i % C, q = i / C;
  const float* p = x + ((size_t)q * 32) * C + c;
  float m = p[0];
  for (int k = 1; k < 32; ++k) m = fmaxf(m, p[(size_t)k * C]);
  out[i] = m;
}

// ---------------------------------------------------------------------------
// flat16[b, c*16 + s] = pts4[b, s, c]   (transpose(0,2,1).reshape, K=8192)
// ---------------------------------------------------------------------------
__global__ void flatten_kernel(const float* __restrict__ pts,
                               _Float16* __restrict__ flat) {
  int i = blockIdx.x * blockDim.x + threadIdx.x;   // over 16*8192
  if (i >= 16 * 8192) return;
  int b = i / 8192, r = i % 8192;
  int c = r >> 4, s = r & 15;
  flat[i] = (_Float16)pts[((size_t)b * 16 + s) * 512 + c];
}

// ---------------------------------------------------------------------------
// Head: train-mode BN over batch (16 rows) + Mish; writes f32 in place and
// the f16 input for the next GEMM (N is already a multiple of 32 here).
// ---------------------------------------------------------------------------
__global__ void bn_mish_kernel(float* __restrict__ x, int N,
                               const float* __restrict__ g,
                               const float* __restrict__ beta,
                               _Float16* __restrict__ out16) {
  int c = blockIdx.x * blockDim.x + threadIdx.x;
  if (c >= N) return;
  float s = 0.f, s2 = 0.f;
  for (int m = 0; m < 16; ++m) {
    float v = x[m * N + c];
    s += v; s2 += v * v;
  }
  float mu  = s / 16.f;
  float var = s2 / 16.f - mu * mu;
  float sc  = g[c] * rsqrtf(var + 1e-5f);
  float sh  = beta[c] - mu * sc;
  for (int m = 0; m < 16; ++m) {
    float v  = x[m * N + c] * sc + sh;
    float sp = log1pf(expf(v));      // softplus; inf-safe (tanh(inf)=1)
    float y  = v * tanhf(sp);
    x[m * N + c]     = y;
    out16[m * N + c] = (_Float16)y;
  }
}

// ---------------------------------------------------------------------------
static void launch_wmma(const _Float16* A16, const _Float16* Wp,
                        const float* bias, float* out, int M, int N, int Kpad,
                        hipStream_t stream) {
  long tiles = (long)(M / 16) * ((N + 15) / 16);
  long threads = tiles * 32;
  int blocks = (int)CDIV(threads, 256);
  wmma_nt_kernel<<<blocks, 256, 0, stream>>>(A16, Wp, bias, out, M, N, Kpad);
}

static void launch_pack_w(const float* w, _Float16* wp, int N, int K,
                          hipStream_t stream) {
  int Npad = align16(N), Kpad = align32(K);
  int tot = Npad * Kpad;
  pack_w_kernel<<<CDIV(tot, 256), 256, 0, stream>>>(w, wp, N, K, Npad, Kpad);
}

extern "C" void kernel_launch(void* const* d_in, const int* in_sizes, int n_in,
                              void* d_out, int out_size, void* d_ws,
                              size_t ws_size, hipStream_t stream) {
  (void)in_sizes; (void)n_in; (void)out_size; (void)ws_size;
  const float* xyz = (const float*)d_in[0];

  // ---- workspace bump allocator (float units; f16 buffers use half slots) --
  float* ws = (float*)d_ws;
  size_t off = 0;
  auto alloc = [&](size_t nfloats) { float* p = ws + off; off += nfloats; return p; };

  const int S[5] = {4096, 1024, 256, 64, 16};
  const int Cfeat[5] = {3, 64, 128, 256, 512};
  float* coords[5];
  coords[0] = alloc((size_t)16 * 4096 * 3);
  for (int l = 1; l < 5; ++l) coords[l] = alloc((size_t)16 * S[l] * 3);
  float* feats[5];
  feats[0] = coords[0];   // level-0 point features == coords
  for (int l = 1; l < 5; ++l) feats[l] = alloc((size_t)16 * S[l] * Cfeat[l]);
  int* fpsidx = (int*)alloc((size_t)16 * 1024);
  int* gidx   = (int*)alloc((size_t)16 * 1024 * 32);
  _Float16* A16 = (_Float16*)alloc((size_t)524288 * 64 / 2);  // 33.5M halves
  float*    F32 = alloc((size_t)524288 * 64);                 // 33.5M f32
  _Float16* WP  = (_Float16*)alloc((size_t)512 * 8192 / 2);   // 4.2M halves
  float* scale = alloc(512);
  float* shift = alloc(512);

  transpose_xyz_kernel<<<CDIV(16 * 4096, 256), 256, 0, stream>>>(xyz, coords[0]);

  const float R2[4] = {0.01f, 0.04f, 0.16f, 0.64f};
  const int mlp[4][4] = {{6, 32, 32, 64}, {67, 64, 64, 128},
                         {131, 128, 128, 256}, {259, 256, 256, 512}};

  for (int l = 0; l < 4; ++l) {
    int n = S[l], np = S[l + 1];
    fps_kernel<<<16, FPS_T, 0, stream>>>(coords[l], n, np, fpsidx);
    int q = 16 * np;
    ballquery_kernel<<<CDIV(q, 256), 256, 0, stream>>>(
        coords[l], fpsidx, R2[l], n, np, coords[l + 1], gidx);
    int M = q * 32;
    int Kpad = align32(mlp[l][0]);
    group_kernel<<<CDIV(M, 256), 256, 0, stream>>>(
        coords[l], feats[l], coords[l + 1], gidx, n, np, Cfeat[l], Kpad, A16);

    int base = 1 + 12 * l;
    for (int j = 0; j < 3; ++j) {
      int K = mlp[l][j];        // logical K (Kpad already set)
      int N = mlp[l][j + 1];
      const float* w  = (const float*)d_in[base + 4 * j + 0];
      const float* bb = (const float*)d_in[base + 4 * j + 1];
      const float* g  = (const float*)d_in[base + 4 * j + 2];
      const float* be = (const float*)d_in[base + 4 * j + 3];
      launch_pack_w(w, WP, N, K, stream);
      launch_wmma(A16, WP, bb, F32, M, N, Kpad, stream);
      bn_stats_kernel<<<N, 256, 0, stream>>>(F32, M, N, g, be, scale, shift);
      int nextKpad = align32(N);
      size_t totPad = (size_t)M * nextKpad;
      bn_relu_f16_kernel<<<(int)CDIV(totPad, 256), 256, 0, stream>>>(
          F32, totPad, N, nextKpad, scale, shift, A16);
      Kpad = nextKpad;
    }
    int C = mlp[l][3];
    size_t pt = (size_t)16 * np * C;
    maxpool_kernel<<<(int)CDIV(pt, 256), 256, 0, stream>>>(
        F32, np, C, feats[l + 1]);
  }

  // ---- classification head ----
  const float* w1  = (const float*)d_in[49];
  const float* b1  = (const float*)d_in[50];
  const float* g1  = (const float*)d_in[51];
  const float* be1 = (const float*)d_in[52];
  const float* w2  = (const float*)d_in[53];
  const float* b2  = (const float*)d_in[54];
  const float* g2  = (const float*)d_in[55];
  const float* be2 = (const float*)d_in[56];
  const float* w3  = (const float*)d_in[57];
  const float* b3  = (const float*)d_in[58];

  flatten_kernel<<<CDIV(16 * 8192, 256), 256, 0, stream>>>(feats[4], A16);
  launch_pack_w(w1, WP, 512, 8192, stream);
  launch_wmma(A16, WP, b1, F32, 16, 512, 8192, stream);
  bn_mish_kernel<<<CDIV(512, 256), 256, 0, stream>>>(F32, 512, g1, be1, A16);
  launch_pack_w(w2, WP, 256, 512, stream);
  launch_wmma(A16, WP, b2, F32, 16, 256, 512, stream);
  bn_mish_kernel<<<1, 256, 0, stream>>>(F32, 256, g2, be2, A16);
  launch_pack_w(w3, WP, 33, 256, stream);
  launch_wmma(A16, WP, b3, (float*)d_out, 16, 33, 256, stream);
}
